// LinearAttention_32152125178316
// MI455X (gfx1250) — matrix operand order repaired and checked by reading, not compiled, not run on hardware
//
#include <hip/hip_runtime.h>
#include <hip/hip_bf16.h>

// ---------------------------------------------------------------------------
// MI455X (gfx1250) fused linear-attention pipeline, bf16 WMMA everywhere.
//   K1: QKV = x @ W^T  (+ RoPE + ReLU epilogue)   -> bf16 workspace
//   K2: flash-style causal ReLU-attention          -> bf16 workspace
//   K3: out = attn_out @ wo^T                      -> fp32 d_out
// GEMM math: v_wmma_f32_16x16x32_bf16, 64x64 wave tiles (16 WMMA / k-step),
// double-buffered LDS (1 barrier per k-step, staging overlaps WMMA),
// async load-to-LDS (ASYNCcnt) for pure bf16 tile copies,
// v_perm_b32 single-op fp32->bf16x2 packing in the staging hot loop.
// ---------------------------------------------------------------------------

typedef __attribute__((ext_vector_type(16))) __bf16 v16bf;
typedef __attribute__((ext_vector_type(8)))  float  v8f;
typedef __attribute__((__vector_size__(4 * sizeof(int)))) int vsi4;

#define SEQ      2048
#define DIM      1024
#define NHEADS   16
#define HEADDIM  64
#define LN_ROPE  9.210340371976184f   // ln(10000)

#if defined(__has_builtin)
#  if __has_builtin(__builtin_amdgcn_global_load_async_to_lds_b128)
#    define HAVE_ASYNC_LDS 1
#  endif
#  if __has_builtin(__builtin_amdgcn_s_wait_asynccnt)
#    define HAVE_WAIT_ASYNC 1
#  endif
#endif

struct frag16 { uint4 lo; uint4 hi; };

__device__ __forceinline__ v16bf as_bf(frag16 f) {
  union { frag16 f; v16bf v; } u; u.f = f; return u.v;
}

// Bulk pack: two fp32 -> two bf16 in ONE v_perm_b32 (truncation; staging path).
__device__ __forceinline__ unsigned pack_bf2(float lo, float hi) {
  union { float f; unsigned u; } a, b; a.f = lo; b.f = hi;
  // pool bytes: S1(=lo) idx 0..3, S0(=hi) idx 4..7; take [hi.b3,hi.b2,lo.b3,lo.b2]
  return __builtin_amdgcn_perm(b.u, a.u, 0x07060302u);
}
__device__ __forceinline__ void store4bf(__bf16* p, float4 f) {
  uint2 u;
  u.x = pack_bf2(f.x, f.y);
  u.y = pack_bf2(f.z, f.w);
  *(uint2*)p = u;
}
// Scalar RNE convert (epilogue quality path).
__device__ __forceinline__ unsigned short f2bf(float x) {
  union { float f; unsigned u; } v; v.f = x;
  unsigned r = v.u + 0x7FFFu + ((v.u >> 16) & 1u);
  return (unsigned short)(r >> 16);
}
__device__ __forceinline__ __bf16 bfval(float x) {
  union { unsigned short u; __bf16 b; } c; c.u = f2bf(x); return c.b;
}
__device__ __forceinline__ v8f vzero() {
  v8f z; for (int i = 0; i < 8; ++i) z[i] = 0.f; return z;
}

// 16-byte global -> LDS copy; async on CDNA5 (ASYNCcnt), sync fallback.
__device__ __forceinline__ void copy16_g2l(__bf16* lds, const __bf16* g) {
#ifdef HAVE_ASYNC_LDS
  __builtin_amdgcn_global_load_async_to_lds_b128(
      (__attribute__((address_space(1))) vsi4*)(void*)g,
      (__attribute__((address_space(3))) vsi4*)(void*)lds, 0, 0);
#else
  *(uint4*)lds = *(const uint4*)g;
#endif
}
__device__ __forceinline__ void async_wait0() {
#ifdef HAVE_ASYNC_LDS
#  ifdef HAVE_WAIT_ASYNC
  __builtin_amdgcn_s_wait_asynccnt(0);
#  else
  asm volatile("s_wait_asynccnt 0" ::: "memory");
#  endif
#endif
}

// A-fragment (16xK rows of M): lane<16 -> row m, k {0..7,16..23}; lane>=16 -> k {8..15,24..31}
__device__ __forceinline__ v16bf lds_fragA(const __bf16* s, int stride, int m0) {
  const int lane = threadIdx.x & 31;
  const __bf16* p = s + (size_t)(m0 + (lane & 15)) * stride + ((lane & 16) ? 8 : 0);
  frag16 f; f.lo = *(const uint4*)p; f.hi = *(const uint4*)(p + 16);
  return as_bf(f);
}
// B-fragment (KxN, stored [n][k] in LDS): lane<16 -> col n, k {0..7,16..23}; lane>=16 -> k {8..15,24..31}
__device__ __forceinline__ v16bf lds_fragB(const __bf16* s, int stride, int n0, int kbase) {
  const int lane = threadIdx.x & 31;
  const __bf16* p = s + (size_t)(n0 + (lane & 15)) * stride + kbase + ((lane & 16) ? 8 : 0);
  frag16 f; f.lo = *(const uint4*)p; f.hi = *(const uint4*)(p + 16);
  return as_bf(f);
}

#define WMMA_BF16(a, b, c) \
  __builtin_amdgcn_wmma_f32_16x16x32_bf16(false, (a), false, (b), (short)0, (c), false, false)

// ---------------------------------------------------------------------------
// Kernel 1: QKV projection + RoPE + ReLU.
// Grid: (32 m-tiles, 12 n-tiles); n-tile 0..3 -> wq, 4..7 -> wk, 8..11 -> wv.
// Block tile 128x256, K-step 32, double-buffered, 64x64 wave tiles (16 WMMA).
// ---------------------------------------------------------------------------
__global__ __launch_bounds__(256)
void qkv_rope_kernel(const float* __restrict__ x,
                     const float* __restrict__ wq,
                     const float* __restrict__ wk,
                     const float* __restrict__ wv,
                     __bf16* __restrict__ qws,   // [b][h][s][d]
                     __bf16* __restrict__ kws,   // [b][h][s][d]
                     __bf16* __restrict__ vtws)  // [b][h][d][s]  (transposed V)
{
  __shared__ __bf16 sA[2][128 * 40];
  __shared__ __bf16 sB[2][256 * 40];

  const int tid  = threadIdx.x;
  const int lane = tid & 31;
  const int wid  = tid >> 5;
  const int m0   = blockIdx.x * 128;
  const int nblk = blockIdx.y;
  const int which = nblk >> 2;                 // 0=Q 1=K 2=V
  const int n0   = (nblk & 3) * 256;           // channel base within weight
  const float* W = (which == 0) ? wq : (which == 1) ? wk : wv;

  const int wm = (wid >> 2) * 64;
  const int wn = (wid & 3) * 64;

  v8f acc[4][4];
  for (int i = 0; i < 4; ++i) for (int j = 0; j < 4; ++j) acc[i][j] = vzero();

  auto stage = [&](int k0, int p) {
    const int kk = (tid & 7) * 4;
    int r = tid >> 3;
    for (int it = 0; it < 4; ++it, r += 32) {
      float4 fx = *(const float4*)(x + (size_t)(m0 + r) * DIM + k0 + kk);
      store4bf(&sA[p][r * 40 + kk], fx);
      if (k0 + 32 < DIM)
        __builtin_prefetch(x + (size_t)(m0 + r) * DIM + k0 + 32 + kk, 0, 3);
    }
    r = tid >> 3;
    for (int it = 0; it < 8; ++it, r += 32) {
      float4 fw = *(const float4*)(W + (size_t)(n0 + r) * DIM + k0 + kk);
      store4bf(&sB[p][r * 40 + kk], fw);
      if (k0 + 32 < DIM)
        __builtin_prefetch(W + (size_t)(n0 + r) * DIM + k0 + 32 + kk, 0, 3);
    }
  };

  stage(0, 0);
  int p = 0;
  for (int k0 = 0; k0 < DIM; k0 += 32, p ^= 1) {
    __syncthreads();                       // buffer p staged & previous readers done
    if (k0 + 32 < DIM) stage(k0 + 32, p ^ 1);   // overlap next stage with WMMAs
    v16bf a[4];
    for (int i = 0; i < 4; ++i) a[i] = lds_fragA(sA[p], 40, wm + i * 16);
    for (int j = 0; j < 4; ++j) {
      v16bf b = lds_fragB(sB[p], 40, wn + j * 16, 0);
      for (int i = 0; i < 4; ++i)
        acc[i][j] = WMMA_BF16(a[i], b, acc[i][j]);
    }
  }

  // Epilogue: RoPE (Q,K) + ReLU, bf16 stores. Frequency hoisted per column.
  for (int j = 0; j < 4; ++j) {
    const int nt = n0 + wn + j * 16 + (lane & 15);  // channel in [0,1024)
    const int h = nt >> 6, d = nt & 63;
    const float freq = __expf(-(float)(d & ~1) * (LN_ROPE / (float)HEADDIM));
    for (int i = 0; i < 4; ++i) {
      for (int r = 0; r < 8; ++r) {
        const int mrow = m0 + wm + i * 16 + r + ((lane >> 4) << 3);
        const int b = mrow >> 11, s = mrow & (SEQ - 1);
        float val = acc[i][j][r];
        if (which < 2) {
          // RoPE: partner channel (d^1) sits in the adjacent lane, same VGPR.
          float other = __shfl_xor(val, 1, 32);
          float sn, cs;
          __sincosf((float)s * freq, &sn, &cs);
          float rot = (d & 1) ? (other * sn + val * cs)
                              : (val * cs - other * sn);
          __bf16 o = bfval(fmaxf(rot, 0.f));             // ReLU
          size_t idx = (((size_t)b * NHEADS + h) * SEQ + s) * HEADDIM + d;
          if (which == 0) qws[idx] = o; else kws[idx] = o;
        } else {
          vtws[(((size_t)b * NHEADS + h) * HEADDIM + d) * SEQ + s] = bfval(val);
        }
      }
    }
  }
}

// ---------------------------------------------------------------------------
// Kernel 2: causal ReLU-attention, flash-style (streaming row-sum).
// Grid: (16 q-tiles, 32 b*h). Block = 8 waves; wave w owns q-rows [q0+16w, +16)
// and the full d=64 output plus per-row denominator.
// K/V tiles: async load-to-LDS, double-buffered.
// ---------------------------------------------------------------------------
__global__ __launch_bounds__(256)
void attn_kernel(const __bf16* __restrict__ qws,
                 const __bf16* __restrict__ kws,
                 const __bf16* __restrict__ vtws,
                 __bf16* __restrict__ aows)   // [b][s][h*64+d]
{
  __shared__ __bf16 sK[2][32 * 72];       // K tile  [key][d]
  __shared__ __bf16 sV[2][64 * 40];       // V^T tile [d][key]
  __shared__ __bf16 sS[8 * 16 * 40];      // per-wave S scratch (D-layout -> A-layout)

  const int tid  = threadIdx.x;
  const int lane = tid & 31;
  const int wid  = tid >> 5;
  const int bh   = blockIdx.y;            // b*16 + h
  const int q0   = blockIdx.x * 128;
  const int qrow0 = q0 + wid * 16;
  const size_t qkbase = (size_t)bh * SEQ * HEADDIM;
  const size_t vtbase = (size_t)bh * HEADDIM * SEQ;

  // Q fragments for this wave's 16 rows (d-chunks {0,32}), straight from global.
  v16bf aq[2];
  {
    const __bf16* qp = qws + qkbase + (size_t)(qrow0 + (lane & 15)) * HEADDIM;
    const int kb = (lane & 16) ? 8 : 0;
    frag16 f;
    f.lo = *(const uint4*)(qp + kb);      f.hi = *(const uint4*)(qp + kb + 16);
    aq[0] = as_bf(f);
    f.lo = *(const uint4*)(qp + 32 + kb); f.hi = *(const uint4*)(qp + 32 + kb + 16);
    aq[1] = as_bf(f);
  }

  v8f acc[4];
  for (int t = 0; t < 4; ++t) acc[t] = vzero();
  float denom[8];
  for (int r = 0; r < 8; ++r) denom[r] = 0.f;

  auto stageKV = [&](int kt0, int p) {
    { int key = tid >> 3, dd = (tid & 7) * 8;
      copy16_g2l(&sK[p][key * 72 + dd],
                 kws + qkbase + (size_t)(kt0 + key) * HEADDIM + dd); }
    { int d = tid >> 2, kk = (tid & 3) * 8;
      copy16_g2l(&sV[p][d * 40 + kk],
                 vtws + vtbase + (size_t)d * SEQ + kt0 + kk); }
  };

  const int kend = q0 + 128;              // causal horizon for this q-tile
  stageKV(0, 0);
  int p = 0;
  for (int kt0 = 0; kt0 < kend; kt0 += 32, p ^= 1) {
    async_wait0();                        // this wave's copies into buffer p done
    __syncthreads();                      // everyone's staging visible
    if (kt0 + 32 < kend) stageKV(kt0 + 32, p ^ 1);

    // S = Q * K^T  (16 x 32 scores, inner dim 64 => 2 WMMA per 16-key column group)
    v8f s0 = vzero(), s1 = vzero();
    for (int kc = 0; kc < 2; ++kc) {
      s0 = WMMA_BF16(aq[kc], lds_fragB(sK[p], 72, 0,  kc * 32), s0);
      s1 = WMMA_BF16(aq[kc], lds_fragB(sK[p], 72, 16, kc * 32), s1);
    }

    // causal mask on diagonal tiles
    if (kt0 + 31 > qrow0) {
      const int k0c = kt0 + (lane & 15);
      for (int r = 0; r < 8; ++r) {
        const int qidx = qrow0 + r + ((lane >> 4) << 3);
        if (k0c      > qidx) s0[r] = 0.f;
        if (k0c + 16 > qidx) s1[r] = 0.f;
      }
    }

    // denominator: row-sum of S (reduce across the 16-lane half)
    for (int r = 0; r < 8; ++r) {
      float t = s0[r] + s1[r];
      t += __shfl_xor(t, 1, 32);
      t += __shfl_xor(t, 2, 32);
      t += __shfl_xor(t, 4, 32);
      t += __shfl_xor(t, 8, 32);
      denom[r] += t;
    }

    // re-layout S (D-layout) -> A-layout via per-wave LDS scratch (same-wave ds order)
    __bf16* ss = sS + wid * 16 * 40;
    for (int r = 0; r < 8; ++r) {
      const int m = r + ((lane >> 4) << 3);
      ss[m * 40 + (lane & 15)]      = bfval(s0[r]);
      ss[m * 40 + 16 + (lane & 15)] = bfval(s1[r]);
    }
    v16bf as = lds_fragA(ss, 40, 0);        // 16 x 32 (keys)

    // out += S * V   (B columns = d, contiguous keys from sV rows)
    for (int t = 0; t < 4; ++t)
      acc[t] = WMMA_BF16(as, lds_fragB(sV[p], 40, t * 16, 0), acc[t]);
  }

  // normalize + store bf16 attn output in [b][s][h*64+d] layout for the final GEMM
  const int b = bh >> 4, h = bh & 15;
  for (int t = 0; t < 4; ++t) {
    const int d = t * 16 + (lane & 15);
    for (int r = 0; r < 8; ++r) {
      const int s = qrow0 + r + ((lane >> 4) << 3);
      float val = acc[t][r] / (denom[r] + 1e-5f);
      aows[((size_t)(b * SEQ + s)) * DIM + h * HEADDIM + d] = bfval(val);
    }
  }
}

// ---------------------------------------------------------------------------
// Kernel 3: final projection out = attn_out @ wo^T, fp32 output.
// Block tile 128x256, double-buffered; A-tile via async LDS copies.
// ---------------------------------------------------------------------------
__global__ __launch_bounds__(256)
void out_proj_kernel(const __bf16* __restrict__ aows,
                     const float* __restrict__ wo,
                     float* __restrict__ out)
{
  __shared__ __bf16 sA[2][128 * 40];
  __shared__ __bf16 sB[2][256 * 40];

  const int tid  = threadIdx.x;
  const int lane = tid & 31;
  const int wid  = tid >> 5;
  const int m0 = blockIdx.x * 128;
  const int n0 = blockIdx.y * 256;
  const int wm = (wid >> 2) * 64;
  const int wn = (wid & 3) * 64;

  v8f acc[4][4];
  for (int i = 0; i < 4; ++i) for (int j = 0; j < 4; ++j) acc[i][j] = vzero();

  auto stage = [&](int k0, int p) {
    { int r = tid >> 1, cc = (tid & 1) * 16;
      copy16_g2l(&sA[p][r * 40 + cc],
                 aows + (size_t)(m0 + r) * DIM + k0 + cc);
      copy16_g2l(&sA[p][r * 40 + cc + 8],
                 aows + (size_t)(m0 + r) * DIM + k0 + cc + 8); }
    { const int kk = (tid & 7) * 4;
      int r = tid >> 3;
      for (int it = 0; it < 8; ++it, r += 32) {
        float4 fw = *(const float4*)(wo + (size_t)(n0 + r) * DIM + k0 + kk);
        store4bf(&sB[p][r * 40 + kk], fw);
        if (k0 + 32 < DIM)
          __builtin_prefetch(wo + (size_t)(n0 + r) * DIM + k0 + 32 + kk, 0, 3);
      } }
  };

  stage(0, 0);
  int p = 0;
  for (int k0 = 0; k0 < DIM; k0 += 32, p ^= 1) {
    async_wait0();
    __syncthreads();
    if (k0 + 32 < DIM) stage(k0 + 32, p ^ 1);

    v16bf a[4];
    for (int i = 0; i < 4; ++i) a[i] = lds_fragA(sA[p], 40, wm + i * 16);
    for (int j = 0; j < 4; ++j) {
      v16bf b = lds_fragB(sB[p], 40, wn + j * 16, 0);
      for (int i = 0; i < 4; ++i)
        acc[i][j] = WMMA_BF16(a[i], b, acc[i][j]);
    }
  }

  for (int i = 0; i < 4; ++i)
    for (int j = 0; j < 4; ++j) {
      const int n = n0 + wn + j * 16 + (lane & 15);
      for (int r = 0; r < 8; ++r) {
        const int m = m0 + wm + i * 16 + r + ((lane >> 4) << 3);
        out[(size_t)m * DIM + n] = acc[i][j][r];
      }
    }
}

// ---------------------------------------------------------------------------
extern "C" void kernel_launch(void* const* d_in, const int* in_sizes, int n_in,
                              void* d_out, int out_size, void* d_ws, size_t ws_size,
                              hipStream_t stream)
{
  const float* x  = (const float*)d_in[0];
  const float* wq = (const float*)d_in[1];
  const float* wk = (const float*)d_in[2];
  const float* wv = (const float*)d_in[3];
  const float* wo = (const float*)d_in[4];

  const size_t per = (size_t)2 * NHEADS * SEQ * HEADDIM;   // 4,194,304 bf16
  __bf16* qws  = (__bf16*)d_ws;
  __bf16* kws  = qws  + per;
  __bf16* vtws = kws  + per;
  __bf16* aows = vtws + per;
  float* out = (float*)d_out;

  qkv_rope_kernel<<<dim3(32, 12), 256, 0, stream>>>(x, wq, wk, wv, qws, kws, vtws);
  attn_kernel   <<<dim3(16, 32), 256, 0, stream>>>(qws, kws, vtws, aows);
  out_proj_kernel<<<dim3(32, 4),  256, 0, stream>>>(aows, wo, out);
}
